// IzhikevichSNN_15264313770388
// MI455X (gfx1250) — compile-verified, matrix operand order checked by reading, not run on hardware
//
#include <hip/hip_runtime.h>

#define T_STEPS 100
#define BATCH   4096
#define IN_RAW  784
#define INP     800          // padded to 25 chunks of 32
#define HID     256
#define NOUT    10
#define KCH     25           // INP / 32
#define BM      128          // batch rows per block (8 waves x 16)
#define BN      64           // hidden cols per block
#define NSL     4            // HID / BN
#define F_MAX_DT 0.1f

typedef _Float16 v16h __attribute__((ext_vector_type(16)));
typedef float    v8f  __attribute__((ext_vector_type(8)));

union V16H { v16h v; _Float16 h[16]; unsigned u[8]; };

// ---------------- CDNA5 async global->LDS copies (ASYNCcnt) ----------------
#define USE_ASYNC 1

__device__ __forceinline__ void cp_async_b128(void* lds, const void* gl) {
#if USE_ASYNC
  unsigned l = (unsigned)(uintptr_t)lds;   // low 32 bits of flat LDS addr = LDS offset
  asm volatile("global_load_async_to_lds_b128 %0, %1, off" :: "v"(l), "v"(gl) : "memory");
#else
  *(uint4*)lds = *(const uint4*)gl;
#endif
}
__device__ __forceinline__ void cp_async_b32(void* lds, const void* gl) {
#if USE_ASYNC
  unsigned l = (unsigned)(uintptr_t)lds;
  asm volatile("global_load_async_to_lds_b32 %0, %1, off" :: "v"(l), "v"(gl) : "memory");
#else
  *(unsigned*)lds = *(const unsigned*)gl;
#endif
}
__device__ __forceinline__ void async_wait0() {
#if USE_ASYNC
  asm volatile("s_wait_asynccnt 0" ::: "memory");
#endif
}

// ---------------- small helpers ----------------
__device__ __forceinline__ unsigned hash_u32(unsigned x) {
  x *= 0x9E3779B9u; x += 0x85EBCA6Bu;
  x ^= x >> 16; x *= 0x7FEB352Du;
  x ^= x >> 15; x *= 0x846CA68Bu;
  x ^= x >> 16; return x;
}
__device__ __forceinline__ unsigned ballot32(bool p) {
#if __has_builtin(__builtin_amdgcn_ballot_w32)
  return __builtin_amdgcn_ballot_w32(p);
#else
  return (unsigned)__ballot(p);
#endif
}

// ---------------- kernel 1: integer thresholds (p * 2^32), padded ----------------
__global__ void thr_kernel(const float* __restrict__ inp, unsigned* __restrict__ thr) {
  int idx = blockIdx.x * 256 + threadIdx.x;
  if (idx >= BATCH * INP) return;
  int b = idx / INP, k = idx - b * INP;
  unsigned v = 0u;
  if (k < IN_RAW) {
    float p = inp[b * IN_RAW + k] * F_MAX_DT;
    v = (unsigned)(p * 4294967296.0f);
  }
  thr[idx] = v;
}

// ---------------- kernel 2: pack fc_w into WMMA B-fragment order (fp16) --------
// wpack[c][nt][lane][e]; B 32x16: lane&15 -> N col, e/lane-hi select K per ISA layout
__global__ void pack_w_kernel(const float* __restrict__ fc_w, _Float16* __restrict__ wpack) {
  int idx = blockIdx.x * 256 + threadIdx.x;
  if (idx >= KCH * 16 * 32 * 16) return;
  int e  = idx & 15;
  int L  = (idx >> 4) & 31;
  int nt = (idx >> 9) & 15;
  int c  = idx >> 13;
  int n  = nt * 16 + (L & 15);
  int kk = ((e < 8) ? e : e + 8) + ((L >> 4) << 3);
  int k  = c * 32 + kk;
  wpack[idx] = (k < IN_RAW) ? (_Float16)fc_w[n * IN_RAW + k] : (_Float16)0.0f;
}

// ---------------- kernel 3: Poisson spike bits, one wave per (t,b) row ---------
__global__ void spike_gen_kernel(const unsigned* __restrict__ thr, unsigned* __restrict__ bits) {
  int lane = threadIdx.x & 31;
  int r = blockIdx.x * 8 + (threadIdx.x >> 5);   // r = t*BATCH + b
  if (r >= T_STEPS * BATCH) return;
  const unsigned* trow = thr + (unsigned)(r & (BATCH - 1)) * INP;
  unsigned base = (unsigned)r * INP;
  #pragma unroll
  for (int c = 0; c < KCH; ++c) {
    int k = c * 32 + lane;
    unsigned tv = trow[k];
    unsigned h  = hash_u32(base + (unsigned)k);
    unsigned m  = ballot32(h < tv);
    if (lane == 0) bits[r * KCH + c] = m;
  }
}

// ---------------- kernel 4: main SNN loop (WMMA + Izhikevich) ------------------
__global__ void __launch_bounds__(256) snn_main_kernel(
    const unsigned* __restrict__ bits_g,     // [T][BATCH][KCH]
    const _Float16* __restrict__ wpack,      // [KCH][16][32][16]
    const float*    __restrict__ fc_b,
    const float*    __restrict__ out_w,      // [NOUT][HID]
    const float*    __restrict__ v0,
    const float*    __restrict__ u0,
    float*          __restrict__ partial)    // [NSL][BATCH][NOUT]
{
  extern __shared__ char smem[];
  _Float16* wlds = (_Float16*)smem;                                 // KCH*4*512 f16
  _Float16* zlds = (_Float16*)(smem + KCH * 4 * 512 * 2);           // BM*BN f16
  unsigned* blds0 = (unsigned*)(smem + KCH * 4 * 512 * 2 + BM * BN * 2);      // [BM][KCH]
  unsigned* blds1 = blds0 + BM * KCH;                                          // double buffer

  const int tid   = threadIdx.x;
  const int lane  = tid & 31;
  const int wv    = tid >> 5;
  const int hi8   = (lane >> 4) << 3;      // 0 or 8
  const int ln    = lane & 15;
  const int mbase = blockIdx.x * BM;
  const int ns    = blockIdx.y;
  const int nbase = ns * BN;
  const int myrow = wv * 16;

  // one-time async copy of this block's weight slice (4 n-tiles x 25 k-chunks)
  for (int i = tid; i < KCH * 4 * 512 / 8; i += 256) {   // 6400 x 16B units
    int cj  = i >> 6;
    int off = (i & 63) * 8;
    int c = cj >> 2, j = cj & 3;
    cp_async_b128(wlds + cj * 512 + off,
                  wpack + ((c * 16 + ns * 4 + j) * 512 + off));
  }
  // prologue: stage bits for t=0 into buffer 0
  {
    const unsigned* src = bits_g + (0 * BATCH + mbase) * KCH;
    for (int i = tid; i < BM * KCH; i += 256) cp_async_b32(blds0 + i, src + i);
  }

  // per-lane constants
  float fcb[4], vi[4], ui[4];
  #pragma unroll
  for (int j = 0; j < 4; ++j) {
    int n = nbase + j * 16 + ln;
    fcb[j] = fc_b[n]; vi[j] = v0[n]; ui[j] = u0[n];
  }
  V16H owf[2];
  #pragma unroll
  for (int c2 = 0; c2 < 2; ++c2) {
    #pragma unroll
    for (int e = 0; e < 16; ++e) {
      int kk = ((e < 8) ? e : e + 8) + hi8;
      int h  = c2 * 32 + kk;
      owf[c2].h[e] = (ln < NOUT) ? (_Float16)out_w[ln * HID + nbase + h] : (_Float16)0.0f;
    }
  }

  float vS[4][8], uS[4][8];
  #pragma unroll
  for (int j = 0; j < 4; ++j)
    #pragma unroll
    for (int g = 0; g < 8; ++g) { vS[j][g] = vi[j]; uS[j][g] = ui[j]; }

  v8f aout;
  #pragma unroll
  for (int g = 0; g < 8; ++g) aout[g] = 0.0f;

  for (int t = 0; t < T_STEPS; ++t) {
    unsigned* bufCur = (t & 1) ? blds1 : blds0;
    unsigned* bufNxt = (t & 1) ? blds0 : blds1;

    // wait for this step's bit tile (and, at t=0, the weight slice); zlds is free
    async_wait0();
    __syncthreads();

    // overlap: issue async DMA of next step's bit tile into the other buffer
    if (t + 1 < T_STEPS) {
      const unsigned* src = bits_g + ((t + 1) * BATCH + mbase) * KCH;
      for (int i = tid; i < BM * KCH; i += 256) cp_async_b32(bufNxt + i, src + i);
    }

    // ---- I = spikes @ fc_w.T via v_wmma_f32_16x16x32_f16 ----
    v8f accI[4];
    #pragma unroll
    for (int j = 0; j < 4; ++j)
      #pragma unroll
      for (int g = 0; g < 8; ++g) accI[j][g] = 0.0f;

    const unsigned* brow = bufCur + (myrow + ln) * KCH;
    for (int c = 0; c < KCH; ++c) {
      unsigned bw = brow[c];
      V16H af;
      #pragma unroll
      for (int e = 0; e < 16; ++e) {
        int kk = ((e < 8) ? e : e + 8) + hi8;
        af.h[e] = ((bw >> kk) & 1u) ? (_Float16)1.0f : (_Float16)0.0f;
      }
      #pragma unroll
      for (int j = 0; j < 4; ++j) {
        v16h bf = *(const v16h*)(wlds + ((c * 4 + j) * 32 + lane) * 16);
        accI[j] = __builtin_amdgcn_wmma_f32_16x16x32_f16(
            false, af.v, false, bf, (short)0, accI[j], false, false);
      }
    }

    // ---- Izhikevich update (state in regs, C-layout) + spike emit to LDS ----
    #pragma unroll
    for (int j = 0; j < 4; ++j) {
      #pragma unroll
      for (int g = 0; g < 8; ++g) {
        float I  = accI[j][g] + fcb[j];
        float v  = vS[j][g], u = uS[j][g];
        float vn = v + 0.25f * (0.04f * v * v + 5.0f * v + 140.0f + I - u);
        float un = u + 0.005f * (0.2f * v - u);
        bool  z  = vn > 30.0f;
        vS[j][g] = z ? -50.0f : vn;
        uS[j][g] = z ? (un + 2.0f) : un;
        zlds[(myrow + g + hi8) * BN + j * 16 + ln] = z ? (_Float16)1.0f : (_Float16)0.0f;
      }
    }
    __syncthreads();

    // ---- acc += z @ out_w.T (WMMA, N padded 10->16, K=64 local) ----
    // (next loop's top barrier protects zlds before it is overwritten)
    #pragma unroll
    for (int c2 = 0; c2 < 2; ++c2) {
      V16H zf;
      const _Float16* zr = zlds + (myrow + ln) * BN + c2 * 32;
      #pragma unroll
      for (int e = 0; e < 16; ++e) {
        int kk = ((e < 8) ? e : e + 8) + hi8;
        zf.h[e] = zr[kk];
      }
      aout = __builtin_amdgcn_wmma_f32_16x16x32_f16(
          false, zf.v, false, owf[c2].v, (short)0, aout, false, false);
    }
  }

  if (ln < NOUT) {
    #pragma unroll
    for (int g = 0; g < 8; ++g) {
      int r = mbase + myrow + g + hi8;
      partial[(ns * BATCH + r) * NOUT + ln] = aout[g];
    }
  }
}

// ---------------- kernel 5: reduce N-slices + T*out_b ----------------
__global__ void finalize_kernel(const float* __restrict__ partial,
                                const float* __restrict__ out_b,
                                float* __restrict__ out) {
  int idx = blockIdx.x * 256 + threadIdx.x;
  if (idx >= BATCH * NOUT) return;
  int o = idx % NOUT;
  float s = (float)T_STEPS * out_b[o];   // reference adds out_b every timestep
  #pragma unroll
  for (int ns = 0; ns < NSL; ++ns) s += partial[ns * BATCH * NOUT + idx];
  out[idx] = s;
}

// ---------------- host launch ----------------
extern "C" void kernel_launch(void* const* d_in, const int* in_sizes, int n_in,
                              void* d_out, int out_size, void* d_ws, size_t ws_size,
                              hipStream_t stream) {
  const float* inp   = (const float*)d_in[0];
  const float* fc_w  = (const float*)d_in[1];
  const float* fc_b  = (const float*)d_in[2];
  const float* out_w = (const float*)d_in[3];
  const float* out_b = (const float*)d_in[4];
  const float* v0    = (const float*)d_in[5];
  const float* u0    = (const float*)d_in[6];

  char* ws = (char*)d_ws;
  const size_t thrBytes   = (size_t)BATCH * INP * 4;                 // 13,107,200
  const size_t bitsBytes  = (size_t)T_STEPS * BATCH * KCH * 4;       // 40,960,000
  const size_t wpackBytes = (size_t)KCH * 16 * 32 * 16 * 2;          //    409,600
  unsigned*  thr     = (unsigned*)ws;
  unsigned*  bits    = (unsigned*)(ws + thrBytes);
  _Float16*  wpack   = (_Float16*)(ws + thrBytes + bitsBytes);
  float*     partial = (float*)(ws + thrBytes + bitsBytes + wpackBytes);

  thr_kernel<<<(BATCH * INP + 255) / 256, 256, 0, stream>>>(inp, thr);
  pack_w_kernel<<<(KCH * 16 * 32 * 16 + 255) / 256, 256, 0, stream>>>(fc_w, wpack);
  spike_gen_kernel<<<T_STEPS * BATCH / 8, 256, 0, stream>>>(thr, bits);

  const size_t smemBytes = (size_t)KCH * 4 * 512 * 2 + (size_t)BM * BN * 2
                         + (size_t)2 * BM * KCH * 4;                 // 144,384 B
  snn_main_kernel<<<dim3(BATCH / BM, NSL), 256, smemBytes, stream>>>(
      bits, wpack, fc_b, out_w, v0, u0, partial);

  finalize_kernel<<<(BATCH * NOUT + 255) / 256, 256, 0, stream>>>(partial, out_b, (float*)d_out);
}
